// IAOAVE2_27530740367439
// MI455X (gfx1250) — matrix operand order repaired and checked by reading, not compiled, not run on hardware
//
#include <hip/hip_runtime.h>
#include <hip/hip_bf16.h>

typedef __attribute__((ext_vector_type(16))) _Float16 v16h;
typedef __attribute__((ext_vector_type(8)))  _Float16 v8h;
typedef __attribute__((ext_vector_type(8)))  float    v8f;

#define BN_ 64
#define SN_ 512
#define HN_ 512
#define DIN_ 1024
#define DINNER_ 1024
#define DS_ 64
#define NH_ 16
#define CONVD_ 1152
#define ZXB_ 2192

#define LDP 40  // LDS row stride in halves: 80B, 16B-aligned, conflict-free frag reads

__device__ __forceinline__ float gelu_tanh(float x) {
  float x3 = x * x * x;
  return 0.5f * x * (1.f + tanhf(0.7978845608028654f * (x + 0.044715f * x3)));
}
__device__ __forceinline__ float silu(float x) { return x / (1.f + expf(-x)); }

// ---------------------------------------------------------------------------
// GEMM: C[M,N] = act(A[M,K] @ W[K,N] + bias), fp32 io, f16 WMMA math.
// 128x128 workgroup tile, 8 waves; wave -> 32x64 of C = 8 x v_wmma per K-step.
// Requires K % 32 == 0 (true for all call sites). M/N edges handled by index
// clamping (loads) + store guards -- fully branchless inner loop.
// ---------------------------------------------------------------------------
__global__ __launch_bounds__(256)
void gemm_f16_wmma(const float* __restrict__ A, const float* __restrict__ W,
                   const float* __restrict__ bias, float* __restrict__ C,
                   int M, int N, int K, int act) {
  __shared__ _Float16 As[128][LDP];   // [m][k]
  __shared__ _Float16 Bs[128][LDP];   // [n][k]  (transposed)
  const int tileN = blockIdx.x * 128;
  const int tileM = blockIdx.y * 128;
  const int tid   = threadIdx.x;
  const int lane  = tid & 31;
  const int wave  = tid >> 5;
  const int wm    = wave & 3;   // 32-row strip
  const int wn    = wave >> 2;  // 64-col strip
  const int lhalf = lane >> 4;
  const int l16   = lane & 15;

  // --- staging assignments (branchless, clamped) ---
  // A: thread -> (row = tid>>1, 16 k's starting at (tid&1)*16); 4 x float4
  const int a_row = tid >> 1;
  const int a_kc  = (tid & 1) * 16;
  const int a_gr  = min(tileM + a_row, M - 1);
  const float* a_base = A + (size_t)a_gr * K + a_kc;
  // B: thread -> (col = tid&127, 16 k's starting at (tid>>7)*16); 16 x b32
  const int b_col = tid & 127;
  const int b_r0  = (tid >> 7) * 16;
  const int b_gc  = min(tileN + b_col, N - 1);

  v8f acc[2][4];
  #pragma unroll
  for (int i = 0; i < 2; ++i)
    #pragma unroll
    for (int j = 0; j < 4; ++j) acc[i][j] = (v8f){};

  float4 a_ld[4];
  float  b_ld[16];

  // prologue: stage k-tile 0
  #pragma unroll
  for (int i = 0; i < 4; ++i) a_ld[i] = ((const float4*)a_base)[i];
  #pragma unroll
  for (int r = 0; r < 16; ++r) b_ld[r] = W[(size_t)(b_r0 + r) * N + b_gc];

  const int ksteps = K >> 5;
  for (int kt = 0; kt < ksteps; ++kt) {
    // ---- stage regs -> LDS (f32 -> f16, two b128 stores each) ----
    {
      v8h p0, p1;
      #pragma unroll
      for (int i = 0; i < 2; ++i) {
        p0[i * 4 + 0] = (_Float16)a_ld[i].x; p0[i * 4 + 1] = (_Float16)a_ld[i].y;
        p0[i * 4 + 2] = (_Float16)a_ld[i].z; p0[i * 4 + 3] = (_Float16)a_ld[i].w;
        p1[i * 4 + 0] = (_Float16)a_ld[i + 2].x; p1[i * 4 + 1] = (_Float16)a_ld[i + 2].y;
        p1[i * 4 + 2] = (_Float16)a_ld[i + 2].z; p1[i * 4 + 3] = (_Float16)a_ld[i + 2].w;
      }
      *(v8h*)&As[a_row][a_kc]     = p0;
      *(v8h*)&As[a_row][a_kc + 8] = p1;
      v8h q0, q1;
      #pragma unroll
      for (int r = 0; r < 8; ++r) { q0[r] = (_Float16)b_ld[r]; q1[r] = (_Float16)b_ld[r + 8]; }
      *(v8h*)&Bs[b_col][b_r0]     = q0;
      *(v8h*)&Bs[b_col][b_r0 + 8] = q1;
    }
    __syncthreads();

    // ---- issue next k-tile's global loads (latency hidden under WMMA) ----
    if (kt + 1 < ksteps) {
      const float* an = a_base + (size_t)(kt + 1) * 32;
      #pragma unroll
      for (int i = 0; i < 4; ++i) a_ld[i] = ((const float4*)an)[i];
      const int kn = (kt + 1) * 32 + b_r0;
      #pragma unroll
      for (int r = 0; r < 16; ++r) b_ld[r] = W[(size_t)(kn + r) * N + b_gc];
    }

    // ---- fragments: pairs of ds_load_b128 at k = 8*lhalf and 16 + 8*lhalf ----
    const int akb = lhalf * 8;
    v16h afr[2];
    #pragma unroll
    for (int mi = 0; mi < 2; ++mi) {
      int row = wm * 32 + mi * 16 + l16;
      v8h lo = *(const v8h*)&As[row][akb];
      v8h hi = *(const v8h*)&As[row][akb + 16];
      afr[mi] = __builtin_shufflevector(lo, hi, 0, 1, 2, 3, 4, 5, 6, 7,
                                        8, 9, 10, 11, 12, 13, 14, 15);
    }
    const int bkb = lhalf * 8;
    v16h bfr[4];
    #pragma unroll
    for (int ni = 0; ni < 4; ++ni) {
      int col = wn * 64 + ni * 16 + l16;
      v8h lo = *(const v8h*)&Bs[col][bkb];
      v8h hi = *(const v8h*)&Bs[col][bkb + 16];
      bfr[ni] = __builtin_shufflevector(lo, hi, 0, 1, 2, 3, 4, 5, 6, 7,
                                        8, 9, 10, 11, 12, 13, 14, 15);
    }
    #pragma unroll
    for (int mi = 0; mi < 2; ++mi)
      #pragma unroll
      for (int ni = 0; ni < 4; ++ni)
        acc[mi][ni] = __builtin_amdgcn_wmma_f32_16x16x32_f16(
            false, afr[mi], false, bfr[ni], (short)0, acc[mi][ni], false, false);
    __syncthreads();
  }

  // ---- epilogue: C/D layout -> row = base + r + 8*lhalf, col = base + l16 ----
  #pragma unroll
  for (int mi = 0; mi < 2; ++mi) {
    int rbase = tileM + wm * 32 + mi * 16 + lhalf * 8;
    #pragma unroll
    for (int ni = 0; ni < 4; ++ni) {
      int col = tileN + wn * 64 + ni * 16 + l16;
      if (col >= N) continue;
      float bv = bias ? bias[col] : 0.f;
      #pragma unroll
      for (int r = 0; r < 8; ++r) {
        int row = rbase + r;
        if (row < M) {
          float v = acc[mi][ni][r] + bv;
          if (act == 1) v = gelu_tanh(v);
          C[(size_t)row * N + col] = v;
        }
      }
    }
  }
}

// ---------------------------------------------------------------------------
// Causal depthwise conv (K=4) over time + bias + SiLU
// ---------------------------------------------------------------------------
__global__ void conv_silu_k(const float* __restrict__ zx, const float* __restrict__ cw,
                            const float* __restrict__ cb, float* __restrict__ out) {
  size_t i = (size_t)blockIdx.x * blockDim.x + threadIdx.x;
  if (i >= (size_t)BN_ * SN_ * CONVD_) return;
  int c = (int)(i % CONVD_);
  int t = (int)((i / CONVD_) % SN_);
  int b = (int)(i / ((size_t)CONVD_ * SN_));
  float acc = cb[c];
  #pragma unroll
  for (int k = 0; k < 4; ++k) {
    int tt = t - 3 + k;
    if (tt >= 0) acc += cw[c * 4 + k] * zx[((size_t)(b * SN_ + tt)) * ZXB_ + DINNER_ + c];
  }
  out[i] = silu(acc);
}

__global__ void dt_decay_k(const float* __restrict__ zx, const float* __restrict__ dtb,
                           const float* __restrict__ alog, float* __restrict__ dt_sp,
                           float* __restrict__ decay) {
  int i = blockIdx.x * blockDim.x + threadIdx.x;
  if (i >= BN_ * SN_ * NH_) return;
  int n = i & (NH_ - 1);
  int bs = i >> 4;
  float v = zx[(size_t)bs * ZXB_ + (DINNER_ + CONVD_) + n] + dtb[n];
  float sp = (v > 20.f) ? v : log1pf(expf(v));
  dt_sp[i] = sp;
  decay[i] = expf(sp * (-expf(alog[n])));
}

// ---------------------------------------------------------------------------
// SSM scan: thread = (b, head n, dim d); DS=64 state in registers; B/C in LDS.
// ---------------------------------------------------------------------------
__global__ __launch_bounds__(256)
void mamba_scan_k(const float* __restrict__ xbc, const float* __restrict__ dt_sp,
                  const float* __restrict__ decay, float* __restrict__ ys) {
  int b = blockIdx.x >> 2;
  int nd = (blockIdx.x & 3) * 256 + threadIdx.x;
  int n = nd >> 6;
  __shared__ float Bs[64], Cs[64];
  float h[64];
  #pragma unroll
  for (int j = 0; j < 64; ++j) h[j] = 0.f;
  for (int s = 0; s < SN_; ++s) {
    const float* base = xbc + (size_t)(b * SN_ + s) * CONVD_;
    if (threadIdx.x < 64) Bs[threadIdx.x] = base[DINNER_ + threadIdx.x];
    else if (threadIdx.x < 128) Cs[threadIdx.x - 64] = base[DINNER_ + DS_ + (threadIdx.x - 64)];
    __syncthreads();
    int idx = (b * SN_ + s) * NH_ + n;
    float dec = decay[idx];
    float dtx = dt_sp[idx] * base[nd];
    float acc = 0.f;
    #pragma unroll
    for (int j = 0; j < 64; ++j) {
      h[j] = h[j] * dec + dtx * Bs[j];
      acc += h[j] * Cs[j];
    }
    ys[(size_t)(b * SN_ + s) * DINNER_ + nd] = acc;
    __syncthreads();
  }
}

__global__ void gate_combine_k(const float* __restrict__ ys, const float* __restrict__ xbc,
                               const float* __restrict__ zx, const float* __restrict__ Dp,
                               float* __restrict__ y) {
  size_t i = (size_t)blockIdx.x * blockDim.x + threadIdx.x;
  if (i >= (size_t)BN_ * SN_ * DINNER_) return;
  int nd = (int)(i & (DINNER_ - 1));
  size_t bs = i >> 10;
  float xs = xbc[bs * CONVD_ + nd];
  float z = zx[bs * ZXB_ + nd];
  y[i] = (ys[i] + Dp[nd >> 6] * xs) * silu(z);
}

__global__ __launch_bounds__(128)
void add_rmsnorm_k(const float* __restrict__ res, float* __restrict__ xo,
                   const float* __restrict__ w) {
  int row = blockIdx.x;
  const float* a = res + (size_t)row * HN_;
  float* o = xo + (size_t)row * HN_;
  __shared__ float red[128];
  float vals[4], loc = 0.f;
  #pragma unroll
  for (int i = 0; i < 4; ++i) {
    int c = threadIdx.x + i * 128;
    float t = a[c] + o[c];
    vals[i] = t; loc += t * t;
  }
  red[threadIdx.x] = loc; __syncthreads();
  for (int st = 64; st > 0; st >>= 1) {
    if (threadIdx.x < st) red[threadIdx.x] += red[threadIdx.x + st];
    __syncthreads();
  }
  float scale = rsqrtf(red[0] / (float)HN_ + 1e-6f);
  #pragma unroll
  for (int i = 0; i < 4; ++i) {
    int c = threadIdx.x + i * 128;
    o[c] = vals[i] * scale * w[c];
  }
}

__global__ __launch_bounds__(256)
void add_layernorm_k(const float* __restrict__ a, const float* __restrict__ badd,
                     const float* __restrict__ g, const float* __restrict__ bvec,
                     float* __restrict__ out, int cols, float eps) {
  int row = blockIdx.x;
  const float* pa = a + (size_t)row * cols;
  const float* pb = badd ? badd + (size_t)row * cols : nullptr;
  float* po = out + (size_t)row * cols;
  __shared__ float rs[256], rs2[256];
  float s = 0.f, s2 = 0.f;
  for (int c = threadIdx.x; c < cols; c += 256) {
    float t = pa[c] + (pb ? pb[c] : 0.f);
    s += t; s2 += t * t;
  }
  rs[threadIdx.x] = s; rs2[threadIdx.x] = s2; __syncthreads();
  for (int st = 128; st > 0; st >>= 1) {
    if (threadIdx.x < st) { rs[threadIdx.x] += rs[threadIdx.x + st]; rs2[threadIdx.x] += rs2[threadIdx.x + st]; }
    __syncthreads();
  }
  float mu = rs[0] / cols;
  float inv = rsqrtf(rs2[0] / cols - mu * mu + eps);
  for (int c = threadIdx.x; c < cols; c += 256) {
    float t = pa[c] + (pb ? pb[c] : 0.f);
    po[c] = (t - mu) * inv * g[c] + bvec[c];
  }
}

// ---------------------------------------------------------------------------
// Attention: one 64-thread block per (b,h,q); scores live in LDS only.
// ---------------------------------------------------------------------------
__global__ __launch_bounds__(64)
void attention_k(const float* __restrict__ qkv, float* __restrict__ out) {
  int idx = blockIdx.x;
  int q = idx & (SN_ - 1);
  int h = (idx >> 9) & 7;
  int b = idx >> 12;
  int tid = threadIdx.x;
  __shared__ float qs[64], sc[SN_], red[64];
  qs[tid] = qkv[((size_t)(b * SN_ + q)) * (3 * HN_) + h * 64 + tid];
  __syncthreads();
  float local[8], m = -1e30f;
  #pragma unroll
  for (int i = 0; i < 8; ++i) {
    int k = tid + i * 64;
    const float4* kp = (const float4*)(qkv + ((size_t)(b * SN_ + k)) * (3 * HN_) + HN_ + h * 64);
    float acc = 0.f;
    #pragma unroll
    for (int d = 0; d < 16; ++d) {
      float4 kv = kp[d];
      acc += qs[d * 4] * kv.x + qs[d * 4 + 1] * kv.y + qs[d * 4 + 2] * kv.z + qs[d * 4 + 3] * kv.w;
    }
    acc *= 0.125f;
    local[i] = acc;
    m = fmaxf(m, acc);
  }
  red[tid] = m; __syncthreads();
  for (int st = 32; st > 0; st >>= 1) { if (tid < st) red[tid] = fmaxf(red[tid], red[tid + st]); __syncthreads(); }
  m = red[0]; __syncthreads();
  float ssum = 0.f;
  #pragma unroll
  for (int i = 0; i < 8; ++i) {
    float p = expf(local[i] - m);
    sc[tid + i * 64] = p;
    ssum += p;
  }
  red[tid] = ssum; __syncthreads();
  for (int st = 32; st > 0; st >>= 1) { if (tid < st) red[tid] += red[tid + st]; __syncthreads(); }
  float inv = 1.f / red[0];
  __syncthreads();
  float acc = 0.f;
  for (int k = 0; k < SN_; ++k)
    acc += sc[k] * qkv[((size_t)(b * SN_ + k)) * (3 * HN_) + 2 * HN_ + h * 64 + tid];
  out[((size_t)(b * SN_ + q)) * HN_ + h * 64 + tid] = acc * inv;
}

__global__ void pool_mean_k(const float* __restrict__ x, float* __restrict__ out, int cols) {
  int i = blockIdx.x * blockDim.x + threadIdx.x;
  if (i >= BN_ * cols) return;
  int b = i / cols, c = i % cols;
  float acc = 0.f;
  for (int s = 0; s < SN_; ++s) acc += x[((size_t)(b * SN_ + s)) * cols + c];
  out[i] = acc * (1.f / SN_);
}

__global__ __launch_bounds__(64)
void router_topk_k(const float* __restrict__ pooled, const float* __restrict__ rw,
                   const float* __restrict__ rb, int* __restrict__ topi, float* __restrict__ topw) {
  int b = blockIdx.x;
  __shared__ float red[64];
  __shared__ float logit[4];
  for (int e = 0; e < 4; ++e) {
    float s = 0.f;
    for (int hh = threadIdx.x; hh < HN_; hh += 64) s += pooled[b * HN_ + hh] * rw[hh * 4 + e];
    red[threadIdx.x] = s; __syncthreads();
    for (int st = 32; st > 0; st >>= 1) { if (threadIdx.x < st) red[threadIdx.x] += red[threadIdx.x + st]; __syncthreads(); }
    if (threadIdx.x == 0) logit[e] = red[0] + rb[e];
    __syncthreads();
  }
  if (threadIdx.x == 0) {
    float mx = fmaxf(fmaxf(logit[0], logit[1]), fmaxf(logit[2], logit[3]));
    float p[4], sum = 0.f;
    for (int e = 0; e < 4; ++e) { p[e] = expf(logit[e] - mx); sum += p[e]; }
    for (int e = 0; e < 4; ++e) p[e] /= sum;
    int i0 = 0;
    for (int e = 1; e < 4; ++e) if (p[e] > p[i0]) i0 = e;
    int i1 = -1;
    for (int e = 0; e < 4; ++e) { if (e == i0) continue; if (i1 < 0 || p[e] > p[i1]) i1 = e; }
    float w0 = p[i0], w1 = p[i1], wsum = w0 + w1;
    topi[b * 2 + 0] = i0; topi[b * 2 + 1] = i1;
    topw[b * 2 + 0] = w0 / wsum; topw[b * 2 + 1] = w1 / wsum;
  }
}

__global__ void moe_combine_k(const float* __restrict__ eo_pooled, const int* __restrict__ topi,
                              const float* __restrict__ topw, float* __restrict__ out) {
  int i = blockIdx.x * blockDim.x + threadIdx.x;
  if (i >= BN_ * 256) return;
  int b = i >> 8, o = i & 255;
  float acc = 0.f;
  #pragma unroll
  for (int k = 0; k < 2; ++k) {
    int e = topi[b * 2 + k];
    acc += topw[b * 2 + k] * eo_pooled[((size_t)e * BN_ + b) * 256 + o];
  }
  out[i] = acc;
}

__global__ __launch_bounds__(64)
void bn_gelu_k(const float* __restrict__ fpre, const float* __restrict__ g,
               const float* __restrict__ bb, float* __restrict__ fout) {
  int j = blockIdx.x;
  int b = threadIdx.x;
  __shared__ float rs[64], rs2[64];
  float v = fpre[b * 256 + j];
  rs[b] = v; rs2[b] = v * v; __syncthreads();
  for (int st = 32; st > 0; st >>= 1) {
    if (b < st) { rs[b] += rs[b + st]; rs2[b] += rs2[b + st]; }
    __syncthreads();
  }
  float mu = rs[0] / 64.f;
  float var = rs2[0] / 64.f - mu * mu;
  float t = (v - mu) * rsqrtf(var + 1e-5f) * g[j] + bb[j];
  fout[b * 256 + j] = gelu_tanh(t);
}

__global__ void fc2_k(const float* __restrict__ f, const float* __restrict__ w,
                      const float* __restrict__ bb, float* __restrict__ out) {
  int i = blockIdx.x * blockDim.x + threadIdx.x;
  if (i >= BN_ * 2) return;
  int b = i >> 1, o = i & 1;
  float acc = bb[o];
  for (int j = 0; j < 256; ++j) acc += f[b * 256 + j] * w[j * 2 + o];
  out[i] = acc;
}

// ---------------------------------------------------------------------------
extern "C" void kernel_launch(void* const* d_in, const int* in_sizes, int n_in,
                              void* d_out, int out_size, void* d_ws, size_t ws_size,
                              hipStream_t stream) {
  (void)in_sizes; (void)n_in; (void)out_size; (void)ws_size;
  const float* x        = (const float*)d_in[0];
  const float* proj_w   = (const float*)d_in[1];
  const float* proj_b   = (const float*)d_in[2];
  const float* m_in_w   = (const float*)d_in[3];
  const float* m_in_b   = (const float*)d_in[4];
  const float* m_conv_w = (const float*)d_in[5];
  const float* m_conv_b = (const float*)d_in[6];
  const float* m_dt_b   = (const float*)d_in[7];
  const float* m_A_log  = (const float*)d_in[8];
  const float* m_D      = (const float*)d_in[9];
  const float* m_out_w  = (const float*)d_in[10];
  const float* m_out_b  = (const float*)d_in[11];
  const float* rms_w    = (const float*)d_in[12];
  const float* qkv_w    = (const float*)d_in[13];
  const float* qkv_b    = (const float*)d_in[14];
  const float* o_w      = (const float*)d_in[15];
  const float* o_b      = (const float*)d_in[16];
  const float* ln1_g    = (const float*)d_in[17];
  const float* ln1_b    = (const float*)d_in[18];
  const float* ffn_w1   = (const float*)d_in[19];
  const float* ffn_b1   = (const float*)d_in[20];
  const float* ffn_w2   = (const float*)d_in[21];
  const float* ffn_b2   = (const float*)d_in[22];
  const float* ln2_g    = (const float*)d_in[23];
  const float* ln2_b    = (const float*)d_in[24];
  const float* router_w = (const float*)d_in[25];
  const float* router_b = (const float*)d_in[26];
  const float* exp_w1   = (const float*)d_in[27];
  const float* exp_b1   = (const float*)d_in[28];
  const float* exp_w2   = (const float*)d_in[29];
  const float* exp_b2   = (const float*)d_in[30];
  const float* fc1_w    = (const float*)d_in[31];
  const float* fc1_b    = (const float*)d_in[32];
  const float* bn1_g    = (const float*)d_in[33];
  const float* bn1_b    = (const float*)d_in[34];
  const float* fc2_w    = (const float*)d_in[35];
  const float* fc2_b    = (const float*)d_in[36];
  const float* on_g     = (const float*)d_in[37];
  const float* on_b     = (const float*)d_in[38];
  float* out = (float*)d_out;

  const size_t MS = (size_t)BN_ * SN_;
  float* ws = (float*)d_ws;
  size_t off = 0;
  float* s1 = ws + off; off += MS * ZXB_;     // zxbcdt -> qkv -> ffn hidden
  float* s2 = ws + off; off += MS * CONVD_;   // conv(xBC) -> expert h1|eo
  float* s3 = ws + off; off += MS * DINNER_;  // ys/y -> attn|oproj -> x4
  float* s4 = ws + off; off += MS * HN_;      // residual -> x3
  float* s5 = ws + off; off += MS * HN_;      // mamba_out/x2 -> ff2
  float* dt_sp      = ws + off; off += MS * NH_;
  float* decay      = ws + off; off += MS * NH_;
  float* pooled     = ws + off; off += (size_t)BN_ * HN_;
  float* eo_pooled  = ws + off; off += (size_t)4 * BN_ * 256;
  float* expert_out = ws + off; off += (size_t)BN_ * 256;
  float* f_pre      = ws + off; off += (size_t)BN_ * 256;
  float* logits_tmp = ws + off; off += 128;
  float* topw       = ws + off; off += 128;
  int*   topi       = (int*)(ws + off); off += 128;

  dim3 blk(256);
  auto g2 = [](int M, int N) { return dim3((unsigned)((N + 127) / 128), (unsigned)((M + 127) / 128)); };
  const int M = (int)MS;

  // 1. proj -> residual (s4)
  gemm_f16_wmma<<<g2(M, HN_), blk, 0, stream>>>(x, proj_w, proj_b, s4, M, HN_, DIN_, 0);
  // 2. mamba in-proj -> zxbcdt (s1)
  gemm_f16_wmma<<<g2(M, ZXB_), blk, 0, stream>>>(s4, m_in_w, m_in_b, s1, M, ZXB_, HN_, 0);
  // 3. causal depthwise conv + silu -> s2
  {
    size_t n = MS * CONVD_;
    conv_silu_k<<<(unsigned)((n + 255) / 256), 256, 0, stream>>>(s1, m_conv_w, m_conv_b, s2);
  }
  // 4. dt softplus + decay
  dt_decay_k<<<(M * NH_ + 255) / 256, 256, 0, stream>>>(s1, m_dt_b, m_A_log, dt_sp, decay);
  // 5. SSM scan -> ys (s3)
  mamba_scan_k<<<BN_ * 4, 256, 0, stream>>>(s2, dt_sp, decay, s3);
  // 6. y = (ys + D*xs)*silu(z)  (in-place s3)
  {
    size_t n = MS * DINNER_;
    gate_combine_k<<<(unsigned)((n + 255) / 256), 256, 0, stream>>>(s3, s2, s1, m_D, s3);
  }
  // 7. mamba out-proj -> s5
  gemm_f16_wmma<<<g2(M, HN_), blk, 0, stream>>>(s3, m_out_w, m_out_b, s5, M, HN_, DINNER_, 0);
  // 8. x2 = rmsnorm(res + mamba_out)  (in-place s5)
  add_rmsnorm_k<<<M, 128, 0, stream>>>(s4, s5, rms_w);
  // 9. qkv -> s1
  gemm_f16_wmma<<<g2(M, 3 * HN_), blk, 0, stream>>>(s5, qkv_w, qkv_b, s1, M, 3 * HN_, HN_, 0);
  // 10. attention -> s3
  attention_k<<<BN_ * 8 * SN_, 64, 0, stream>>>(s1, s3);
  // 11. output proj -> s3 + MS*HN_
  gemm_f16_wmma<<<g2(M, HN_), blk, 0, stream>>>(s3, o_w, o_b, s3 + MS * HN_, M, HN_, HN_, 0);
  // 12. x3 = ln1(x2 + oproj) -> s4
  add_layernorm_k<<<M, 256, 0, stream>>>(s5, s3 + MS * HN_, ln1_g, ln1_b, s4, HN_, 1e-5f);
  // 13. ffn1 (+gelu) -> s1
  gemm_f16_wmma<<<g2(M, 2048), blk, 0, stream>>>(s4, ffn_w1, ffn_b1, s1, M, 2048, HN_, 1);
  // 14. ffn2 -> s5
  gemm_f16_wmma<<<g2(M, HN_), blk, 0, stream>>>(s1, ffn_w2, ffn_b2, s5, M, HN_, 2048, 0);
  // 15. x4 = ln2(x3 + ff2) -> s3
  add_layernorm_k<<<M, 256, 0, stream>>>(s4, s5, ln2_g, ln2_b, s3, HN_, 1e-5f);
  // 16. pooled = mean_s(x4)
  pool_mean_k<<<(BN_ * HN_ + 255) / 256, 256, 0, stream>>>(s3, pooled, HN_);
  // 17. router softmax + top2
  router_topk_k<<<BN_, 64, 0, stream>>>(pooled, router_w, router_b, topi, topw);
  // 18. experts
  for (int e = 0; e < 4; ++e) {
    gemm_f16_wmma<<<g2(M, 256), blk, 0, stream>>>(s3, exp_w1 + (size_t)e * HN_ * 256, exp_b1 + e * 256,
                                                  s2, M, 256, HN_, 1);
    gemm_f16_wmma<<<g2(M, 256), blk, 0, stream>>>(s2, exp_w2 + (size_t)e * 256 * 256, exp_b2 + e * 256,
                                                  s2 + MS * 256, M, 256, 256, 0);
    pool_mean_k<<<(BN_ * 256 + 255) / 256, 256, 0, stream>>>(s2 + MS * 256, eo_pooled + (size_t)e * BN_ * 256, 256);
  }
  // 19. top-2 combine
  moe_combine_k<<<(BN_ * 256 + 255) / 256, 256, 0, stream>>>(eo_pooled, topi, topw, expert_out);
  // 20. fc1
  gemm_f16_wmma<<<g2(BN_, 256), blk, 0, stream>>>(expert_out, fc1_w, fc1_b, f_pre, BN_, 256, 256, 0);
  // 21. batchnorm + gelu -> f (second output)
  bn_gelu_k<<<256, 64, 0, stream>>>(f_pre, bn1_g, bn1_b, out + BN_ * 2);
  // 22. fc2 -> logits_tmp
  fc2_k<<<1, 128, 0, stream>>>(out + BN_ * 2, fc2_w, fc2_b, logits_tmp);
  // 23. final layernorm over 2 -> logits (first output)
  add_layernorm_k<<<BN_, 256, 0, stream>>>(logits_tmp, nullptr, on_g, on_b, out, 2, 1e-5f);
}
